// LinearSelfAttention_77197742178887
// MI455X (gfx1250) — matrix operand order repaired and checked by reading, not compiled, not run on hardware
//
#include <hip/hip_runtime.h>
#include <hip/hip_bf16.h>

// ---- problem constants (from reference) ----
#define B_  4
#define C_  256
#define H_  128
#define W_  128
#define N_  (H_*W_)      // 16384
#define EPS_ 1e-10f

typedef _Float16 h16 __attribute__((ext_vector_type(16)));
typedef float    f8  __attribute__((ext_vector_type(8)));
typedef float    f4  __attribute__((ext_vector_type(4)));

union FragH { h16 v; f4 q[2]; _Float16 e[16]; };
union H8    { f4 v;  _Float16 e[8]; };

#define XSTRIDE 40   // padded LDS stride (halves) for transposed X tile

__device__ __forceinline__ float delu_f(float x) {
    // delu(x) = x>=0 ? 10x+1 : exp(10x)
    return x >= 0.f ? __builtin_fmaf(10.f, x, 1.f) : __expf(10.f * x);
}

__device__ __forceinline__ f8 wmma_f16(h16 a, h16 b, f8 c) {
    return __builtin_amdgcn_wmma_f32_16x16x32_f16(false, a, false, b, (short)0, c, false, false);
}

// CDNA5 async global->LDS copy (16B per lane), tracked by ASYNCcnt.
__device__ __forceinline__ void async_copy_b128(void* lds_dst, const void* gsrc) {
    unsigned int l = (unsigned int)(unsigned long long)lds_dst; // low 32 bits of the pointer give the LDS address
    asm volatile("global_load_async_to_lds_b128 %0, %1, off" :: "v"(l), "v"(gsrc) : "memory");
}
__device__ __forceinline__ void async_wait0() {
    asm volatile("s_wait_asynccnt 0x0" ::: "memory");
}

// A fragment: 16(M)x32(K), source row-major, ld in halves (multiple of 8).
// Lanes 0-15: M=lane, K 0..7 & 16..23 ; lanes 16-31: M=lane-16, K 8..15 & 24..31.
__device__ __forceinline__ h16 load_frag_a(const _Float16* p, int ld) {
    int lane = threadIdx.x & 31;
    int m  = lane & 15;
    int ko = (lane >> 4) * 8;
    FragH f;
    f.q[0] = *(const f4*)(p + m * ld + ko);
    f.q[1] = *(const f4*)(p + m * ld + 16 + ko);
    return f.v;
}

// B fragment from TRANSPOSED source: src[n*ld + k] = B[k][n]; two 16B loads per lane.
// Lanes 0-15: N=lane, K 0..7 & 16..23 ; lanes 16-31: N=lane-16, K 8..15 & 24..31.
__device__ __forceinline__ h16 load_frag_bt(const _Float16* p, int ld) {
    int lane = threadIdx.x & 31;
    int n  = lane & 15;
    int ko = (lane >> 4) * 8;
    FragH f;
    f.q[0] = *(const f4*)(p + n * ld + ko);
    f.q[1] = *(const f4*)(p + n * ld + 16 + ko);
    return f.v;
}

// ---------------- K0: conversions ----------------
__global__ void cast_f16_k(const float* __restrict__ s, _Float16* __restrict__ d, int n) {
    int i = blockIdx.x * 256 + threadIdx.x;
    if (i < n) d[i] = (_Float16)s[i];
}

// Build 9-tap weights [tap][o][i] in f16; center tap (tap==4) absorbs wd1 (conv1x1).
__global__ void build_w3_k(const float* __restrict__ wd3, const float* __restrict__ wd1,
                           _Float16* __restrict__ w3h) {
    int i = blockIdx.x * 256 + threadIdx.x;          // 9*65536 total
    int tap = i >> 16, r = i & 65535;
    int o = r >> 8, ic = r & 255;
    float v = wd3[o * 2304 + ic * 9 + tap];
    if (tap == 4) v += wd1[r];
    w3h[i] = (_Float16)v;
}

__global__ void zero_f32_k(float* p, int n) {
    int i = blockIdx.x * 256 + threadIdx.x;
    if (i < n) p[i] = 0.f;
}

// ---------------- K1: fused q/k/v GEMM (stacked 768x256 weights) ----------------
// grid (N/64, 6, B), block 256. Block tile: 128(M) x 64(N), K=256 in steps of 32.
// Double-buffered LDS: async W-tile DMA overlaps WMMA compute; one barrier per K-step.
__global__ __launch_bounds__(256) void qkv_gemm_k(
    const _Float16* __restrict__ xh, const _Float16* __restrict__ wqkv,
    const float* __restrict__ bq, const float* __restrict__ bk, const float* __restrict__ bv,
    _Float16* __restrict__ qh, _Float16* __restrict__ kh, _Float16* __restrict__ vh)
{
    __shared__ _Float16 WT[2][128 * 32];     // W tile, row-major (M x K)
    __shared__ _Float16 XT[2][64 * XSTRIDE]; // X tile, TRANSPOSED (n x K), padded
    int t = threadIdx.x, wave = t >> 5, lane = t & 31;
    int b = blockIdx.z, by = blockIdx.y, n0 = blockIdx.x * 64;
    int M0 = by * 128;
    int wm = wave & 1, wn = wave >> 1;        // 2x4 wave grid
    f8 acc0 = {}, acc1 = {}, acc2 = {}, acc3 = {};
    const _Float16* xb = xh + (size_t)b * C_ * N_;

    auto stage = [&](int kk, int buf) {
        int k0 = kk * 32;
        {   // async-DMA W tile 128x32 into LDS (2 x 16B per thread)
            int row = t >> 1, cb = (t & 1) * 16;
            const _Float16* src = wqkv + (size_t)(M0 + row) * 256 + k0 + cb;
            async_copy_b128(&WT[buf][row * 32 + cb],     src);
            async_copy_b128(&WT[buf][row * 32 + cb + 8], src + 8);
        }
        {   // X tile 32(K)x64(n), stored transposed into XT[n][K]
            int row = t >> 3, cb = (t & 7) * 8;
            H8 tmp;
            tmp.v = *(const f4*)(xb + (size_t)(k0 + row) * N_ + n0 + cb);
#pragma unroll
            for (int j = 0; j < 8; ++j) XT[buf][(cb + j) * XSTRIDE + row] = tmp.e[j];
        }
    };

    stage(0, 0);
    for (int kk = 0; kk < 8; ++kk) {
        int cur = kk & 1, nxt = cur ^ 1;
        async_wait0();          // W tile for 'cur' complete
        __syncthreads();        // X stores visible; prior readers of 'nxt' done
        if (kk < 7) stage(kk + 1, nxt);
        if (kk < 6) {           // distance-2 prefetch of the X stream
            int row = t >> 3, cb = (t & 7) * 8;
            __builtin_prefetch(xb + (size_t)((kk + 2) * 32 + row) * N_ + n0 + cb, 0, 1);
        }
        h16 bfrag = load_frag_bt(&XT[cur][(wn * 16) * XSTRIDE], XSTRIDE);
        h16 a0 = load_frag_a(&WT[cur][(wm * 64 +  0) * 32], 32);
        h16 a1 = load_frag_a(&WT[cur][(wm * 64 + 16) * 32], 32);
        h16 a2 = load_frag_a(&WT[cur][(wm * 64 + 32) * 32], 32);
        h16 a3 = load_frag_a(&WT[cur][(wm * 64 + 48) * 32], 32);
        acc0 = wmma_f16(a0, bfrag, acc0);
        acc1 = wmma_f16(a1, bfrag, acc1);
        acc2 = wmma_f16(a2, bfrag, acc2);
        acc3 = wmma_f16(a3, bfrag, acc3);
    }

    // epilogue: whole block belongs to one of q/k/v (128-row tile within a 256-row matrix)
    int mat = by >> 1;
    int cb0 = (by & 1) * 128 + wm * 64;
    const float* bias = (mat == 0) ? bq : (mat == 1) ? bk : bv;
    _Float16* dst = (mat == 0) ? qh : (mat == 1) ? kh : vh;
    int col = lane & 15, rbase = (lane >> 4) * 8;
    f8 accs[4] = {acc0, acc1, acc2, acc3};
#pragma unroll
    for (int s = 0; s < 4; ++s) {
#pragma unroll
        for (int r = 0; r < 8; ++r) {
            int c_out = cb0 + s * 16 + rbase + r;
            float v = accs[s][r] + bias[c_out];
            if (mat < 2) v = delu_f(v);
            dst[((size_t)b * C_ + c_out) * N_ + n0 + wn * 16 + col] = (_Float16)v;
        }
    }
}

// ---------------- K2: kv = delu(k) * v (in place over v buffer) ----------------
__global__ void kv_mul_k(const _Float16* __restrict__ kh, _Float16* __restrict__ vh, int n) {
    int i = blockIdx.x * 256 + threadIdx.x;
    if (i < n) vh[i] = (_Float16)((float)kh[i] * (float)vh[i]);
}

// ---------------- K3: energy = qh @ kh^T, split-K over N with f32 atomics ----------------
// grid (32, 8, B), block 256 (8 waves). Each wave owns one 16x16 (m,c) tile over a 2048 chunk.
__global__ __launch_bounds__(256) void energy_gemm_k(
    const _Float16* __restrict__ qh, const _Float16* __restrict__ kh, float* __restrict__ E)
{
    int t = threadIdx.x, wave = t >> 5, lane = t & 31;
    int b = blockIdx.z;
    int tile = blockIdx.x * 8 + wave;         // 0..255
    int m0 = (tile >> 4) * 16, c0 = (tile & 15) * 16;
    int nbeg = blockIdx.y * (N_ / 8);
    f8 acc = {};
    const _Float16* qrow = qh + ((size_t)b * C_ + m0 + (lane & 15)) * N_;
    const _Float16* krow = kh + ((size_t)b * C_ + c0 + (lane & 15)) * N_;
    int aoff = (lane >> 4) * 8;
    int boff = (lane >> 4) * 8;
    for (int k0 = nbeg; k0 < nbeg + N_ / 8; k0 += 32) {
        FragH a, bf;
        a.q[0]  = *(const f4*)(qrow + k0 + aoff);
        a.q[1]  = *(const f4*)(qrow + k0 + 16 + aoff);
        bf.q[0] = *(const f4*)(krow + k0 + boff);
        bf.q[1] = *(const f4*)(krow + k0 + 16 + boff);
        acc = wmma_f16(a.v, bf.v, acc);
    }
    int col = lane & 15, rbase = (lane >> 4) * 8;
    float* Eb = E + (size_t)b * 65536;
#pragma unroll
    for (int r = 0; r < 8; ++r)
        atomicAdd(Eb + (size_t)(m0 + rbase + r) * 256 + c0 + col, acc[r]);
}

// ---------------- K4: evec[c] = sum_m E[m,c] ----------------
__global__ void col_sum_k(const float* __restrict__ E, float* __restrict__ evec) {
    int b = blockIdx.x, c = threadIdx.x;
    const float* Eb = E + (size_t)b * 65536;
    float s = 0.f;
    for (int m = 0; m < 256; ++m) s += Eb[(size_t)m * 256 + c];
    evec[b * 256 + c] = s;
}

// ---------------- K5: implicit-GEMM 3x3 conv (+fused 1x1 d1), writes out = x + g*lin -------
// grid (N/64, 2, B), block 256. K total = (valid taps) * 256, double-buffered pipeline.
__global__ __launch_bounds__(256) void lin_conv_k(
    const _Float16* __restrict__ xh, const _Float16* __restrict__ w3h,
    const float* __restrict__ x, const float* __restrict__ bd1, const float* __restrict__ bd3,
    const float* __restrict__ gamma, float* __restrict__ out)
{
    __shared__ _Float16 WT[2][128 * 32];
    __shared__ _Float16 XT[2][64 * XSTRIDE]; // transposed (n x K)
    int t = threadIdx.x, wave = t >> 5, lane = t & 31;
    int b = blockIdx.z, by = blockIdx.y, n0 = blockIdx.x * 64;
    int obase = by * 128;
    int h = n0 >> 7, w0 = n0 & 127;
    int wm = wave & 1, wn = wave >> 1;
    f8 acc0 = {}, acc1 = {}, acc2 = {}, acc3 = {};
    const _Float16* xb = xh + (size_t)b * C_ * N_;

    // valid taps for this block (uniform): dy in range
    int vtap[9], nv = 0;
    for (int tap = 0; tap < 9; ++tap) {
        int h2 = h + tap / 3 - 1;
        if (h2 >= 0 && h2 < H_) vtap[nv++] = tap;
    }
    int nsteps = nv * 8;

    auto stage = [&](int step, int buf) {
        int tap = vtap[step >> 3];
        int k0 = (step & 7) * 32;
        int dy = tap / 3 - 1, dx = tap % 3 - 1;
        int h2 = h + dy;
        {   // async-DMA W tap tile 128x32
            int row = t >> 1, cb = (t & 1) * 16;
            const _Float16* src = w3h + (size_t)tap * 65536 + (size_t)(obase + row) * 256 + k0 + cb;
            async_copy_b128(&WT[buf][row * 32 + cb],     src);
            async_copy_b128(&WT[buf][row * 32 + cb + 8], src + 8);
        }
        {   // shifted X tile with SAME-padding zeros, stored transposed
            int row = t >> 3, cb = (t & 7) * 8;
#pragma unroll
            for (int j = 0; j < 8; ++j) {
                int w2 = w0 + cb + j + dx;
                _Float16 v = (_Float16)0.f;
                if (w2 >= 0 && w2 < W_)
                    v = xb[(size_t)(k0 + row) * N_ + h2 * W_ + w2];
                XT[buf][(cb + j) * XSTRIDE + row] = v;
            }
        }
    };

    stage(0, 0);
    for (int step = 0; step < nsteps; ++step) {
        int cur = step & 1, nxt = cur ^ 1;
        async_wait0();
        __syncthreads();
        if (step + 1 < nsteps) stage(step + 1, nxt);
        h16 bfrag = load_frag_bt(&XT[cur][(wn * 16) * XSTRIDE], XSTRIDE);
        h16 a0 = load_frag_a(&WT[cur][(wm * 64 +  0) * 32], 32);
        h16 a1 = load_frag_a(&WT[cur][(wm * 64 + 16) * 32], 32);
        h16 a2 = load_frag_a(&WT[cur][(wm * 64 + 32) * 32], 32);
        h16 a3 = load_frag_a(&WT[cur][(wm * 64 + 48) * 32], 32);
        acc0 = wmma_f16(a0, bfrag, acc0);
        acc1 = wmma_f16(a1, bfrag, acc1);
        acc2 = wmma_f16(a2, bfrag, acc2);
        acc3 = wmma_f16(a3, bfrag, acc3);
    }

    float g = gamma[0];
    int col = lane & 15, rbase = (lane >> 4) * 8;
    f8 accs[4] = {acc0, acc1, acc2, acc3};
#pragma unroll
    for (int s2 = 0; s2 < 4; ++s2)
#pragma unroll
        for (int r = 0; r < 8; ++r) {
            int o = obase + wm * 64 + s2 * 16 + rbase + r;
            size_t idx = ((size_t)b * C_ + o) * N_ + n0 + wn * 16 + col;
            out[idx] = x[idx] + g * (accs[s2][r] + bd1[o] + bd3[o]);
        }
}

// ---------------- K6: norm + attention, out += g * evec[c]*kv[c,n]*norm[n] ----------------
__global__ __launch_bounds__(256) void attn_out_k(
    const _Float16* __restrict__ kh, const _Float16* __restrict__ kvh,
    const float* __restrict__ evec, const float* __restrict__ gamma,
    float* __restrict__ out)
{
    __shared__ float se[256];
    int t = threadIdx.x, b = blockIdx.y;
    int n = blockIdx.x * 256 + t;
    se[t] = evec[b * 256 + t];
    __syncthreads();
    const _Float16* khb = kh + (size_t)b * C_ * N_;
    float s = 0.f;
    for (int c = 0; c < 256; ++c) s += se[c] * (float)khb[(size_t)c * N_ + n];
    float nr = 1.f / (s + EPS_);
    float g = gamma[0];
    const _Float16* kvb = kvh + (size_t)b * C_ * N_;
    float* ob = out + (size_t)b * C_ * N_;
    for (int c = 0; c < 256; ++c) {
        size_t idx = (size_t)c * N_ + n;
        ob[idx] += g * se[c] * (float)kvb[idx] * nr;
    }
}

// ---------------- launcher ----------------
extern "C" void kernel_launch(void* const* d_in, const int* in_sizes, int n_in,
                              void* d_out, int out_size, void* d_ws, size_t ws_size,
                              hipStream_t stream) {
    const float* x    = (const float*)d_in[0];
    const float* wq   = (const float*)d_in[1];
    const float* bq   = (const float*)d_in[2];
    const float* wk   = (const float*)d_in[3];
    const float* bk   = (const float*)d_in[4];
    const float* wv   = (const float*)d_in[5];
    const float* bv   = (const float*)d_in[6];
    const float* wd1  = (const float*)d_in[7];
    const float* bd1  = (const float*)d_in[8];
    const float* wd3  = (const float*)d_in[9];
    const float* bd3  = (const float*)d_in[10];
    const float* gma  = (const float*)d_in[11];
    float* out = (float*)d_out;

    const size_t BCN   = (size_t)B_ * C_ * N_;      // 16777216
    char* ws = (char*)d_ws;
    size_t o = 0;
    auto take = [&](size_t bytes) { char* p = ws + o; o += (bytes + 255) & ~(size_t)255; return p; };
    _Float16* xh   = (_Float16*)take(BCN * 2);
    _Float16* qh   = (_Float16*)take(BCN * 2);
    _Float16* kh   = (_Float16*)take(BCN * 2);
    _Float16* vh   = (_Float16*)take(BCN * 2);          // v, then kv in place
    _Float16* wqkv = (_Float16*)take((size_t)768 * 256 * 2);
    _Float16* w3h  = (_Float16*)take((size_t)9 * 65536 * 2);
    float*    E    = (float*)take((size_t)B_ * 65536 * 4);
    float*    evec = (float*)take((size_t)B_ * 256 * 4);

    // K0: conversions
    cast_f16_k<<<(int)(BCN / 256), 256, 0, stream>>>(x, xh, (int)BCN);
    cast_f16_k<<<256, 256, 0, stream>>>(wq, wqkv,           65536);
    cast_f16_k<<<256, 256, 0, stream>>>(wk, wqkv + 65536,   65536);
    cast_f16_k<<<256, 256, 0, stream>>>(wv, wqkv + 131072,  65536);
    build_w3_k<<<2304, 256, 0, stream>>>(wd3, wd1, w3h);
    zero_f32_k<<<1024, 256, 0, stream>>>(E, B_ * 65536);

    // K1: q/k/v projections + delu
    qkv_gemm_k<<<dim3(N_ / 64, 6, B_), 256, 0, stream>>>(xh, wqkv, bq, bk, bv, qh, kh, vh);

    // K2: kv = k*v
    kv_mul_k<<<(int)(BCN / 256), 256, 0, stream>>>(kh, vh, (int)BCN);

    // K3/K4: energy and its column sums
    energy_gemm_k<<<dim3(32, 8, B_), 256, 0, stream>>>(qh, kh, E);
    col_sum_k<<<B_, 256, 0, stream>>>(E, evec);

    // K5: linear path (1x1 d + 3x3 conv) -> out = x + g*lin
    lin_conv_k<<<dim3(N_ / 64, 2, B_), 256, 0, stream>>>(xh, w3h, x, bd1, bd3, gma, out);

    // K6: out += g * attention
    attn_out_k<<<dim3(N_ / 256, B_), 256, 0, stream>>>(kh, vh, evec, gma, out);
}
